// SpikingNet_61151744361016
// MI455X (gfx1250) — matrix operand order repaired and checked by reading, not compiled, not run on hardware
//
#include <hip/hip_runtime.h>

// ---------------------------------------------------------------------------
// SNN forward: 25 steps of {GEMM1 + LIF, GEMM2 + LIF} on gfx1250 (wave32 WMMA)
// GEMM1 stages both tiles with GLOBAL_LOAD_ASYNC_TO_LDS_B128 (ASYNCcnt path).
// K-loop unroll capped at 2 to keep staging addresses incremental (occupancy).
// ---------------------------------------------------------------------------

typedef __attribute__((ext_vector_type(16))) _Float16 v16h;
typedef __attribute__((ext_vector_type(8)))  _Float16 v8h;
typedef __attribute__((ext_vector_type(4)))  _Float16 v4h;
typedef __attribute__((ext_vector_type(8)))  float    v8f;

union Frag { v16h v; uint4 q[2]; };

#define BETA 0.95f
#define THR  1.0f

// Problem sizes
#define NSTEP 25
#define B     4096
#define NIN   784
#define NHID  1000
#define NOUT  10
// Padded sizes
#define KP    800      // 25 * 32
#define HP    1024     // 64 * 16
#define OP16  16

// ---------------------------------------------------------------------------
// Prep kernels (run every launch; ws is poisoned by the harness)
// ---------------------------------------------------------------------------
__global__ void zero_f32_kernel(float* __restrict__ p, int n) {
    int i = blockIdx.x * blockDim.x + threadIdx.x;
    if (i < n) p[i] = 0.0f;
}

__global__ void prep_w1_kernel(const float* __restrict__ W1, _Float16* __restrict__ W1h) {
    int i = blockIdx.x * blockDim.x + threadIdx.x;      // HP * KP
    if (i >= HP * KP) return;
    int h = i / KP, k = i - h * KP;
    float v = (h < NHID && k < NIN) ? W1[h * NIN + k] : 0.0f;
    W1h[i] = (_Float16)v;
}

__global__ void prep_w2_kernel(const float* __restrict__ W2, _Float16* __restrict__ W2h) {
    int i = blockIdx.x * blockDim.x + threadIdx.x;      // OP16 * HP
    if (i >= OP16 * HP) return;
    int o = i >> 10, h = i & (HP - 1);
    float v = (o < NOUT && h < NHID) ? W2[o * NHID + h] : 0.0f;
    W2h[i] = (_Float16)v;
}

__global__ void prep_b1_kernel(const float* __restrict__ b1, float* __restrict__ b1p) {
    int i = blockIdx.x * blockDim.x + threadIdx.x;
    if (i < HP) b1p[i] = (i < NHID) ? b1[i] : 0.0f;
}

// Per-step: x_t (f32) -> Xh (f16, K zero-padded to 800). Each thread: 4 elems.
__global__ void prep_x_kernel(const float* __restrict__ x, _Float16* __restrict__ Xh) {
    int i = blockIdx.x * blockDim.x + threadIdx.x;      // B * KP/4
    if (i >= B * (KP / 4)) return;
    int row = i / (KP / 4);
    int k   = (i - row * (KP / 4)) * 4;                 // multiple of 4; 784%4==0
    v4h h;
    if (k < NIN) {
        float4 f = *(const float4*)(x + (size_t)row * NIN + k);
        h[0] = (_Float16)f.x; h[1] = (_Float16)f.y;
        h[2] = (_Float16)f.z; h[3] = (_Float16)f.w;
    } else {
        h[0] = h[1] = h[2] = h[3] = (_Float16)0.0f;
    }
    *(v4h*)(Xh + (size_t)row * KP + k) = h;
}

// ---------------------------------------------------------------------------
// GEMM1 + LIF layer 1 :  cur1 = Xh @ W1h^T ;  mem1/spk1 update fused
//   block tile 128(M) x 128(N), 8 waves, each wave 32x64 (2x4 wmma tiles)
//   K loop: 25 x 32, double-buffered LDS filled by async-to-LDS copies
// ---------------------------------------------------------------------------
#define BM  128
#define BN  128
#define BK  32
#define LDT 40   // LDS row stride in halfs (32 + 8 pad, keeps 16B alignment)
#define NKT (KP / BK)

__global__ __launch_bounds__(256)
void gemm1_lif_kernel(const _Float16* __restrict__ Xh,      // [B, KP]  (step t)
                      const _Float16* __restrict__ W1h,     // [HP, KP]
                      const float* __restrict__ b1p,        // [HP]
                      float* __restrict__ mem1,             // [B, HP]
                      _Float16* __restrict__ spk1)          // [B, HP]
{
    __shared__ _Float16 As[2][BM * LDT];
    __shared__ _Float16 Bs[2][BN * LDT];

    const int tid  = threadIdx.x;
    const int lane = tid & 31;
    const int wid  = tid >> 5;
    const int wm   = wid & 3;          // 4 wave rows  -> 32 M rows each
    const int wn   = wid >> 2;         // 2 wave cols  -> 64 N cols each
    const int mBase = blockIdx.y * BM;
    const int nBase = blockIdx.x * BN;

    // Cooperative staging: 2 threads per row, 32 bytes (16 halfs) each
    const int aRow = tid >> 1;
    const int aCol = (tid & 1) << 4;   // 0 or 16 (halfs)
    const _Float16* Xrow = Xh  + (size_t)(mBase + aRow) * KP + aCol;
    const _Float16* Wrow = W1h + (size_t)(nBase + aRow) * KP + aCol;

    // LDS byte offsets of this thread's staging destination, per buffer
    const unsigned dA[2] = {
        (unsigned)(size_t)(&As[0][aRow * LDT + aCol]),
        (unsigned)(size_t)(&As[1][aRow * LDT + aCol]) };
    const unsigned dB[2] = {
        (unsigned)(size_t)(&Bs[0][aRow * LDT + aCol]),
        (unsigned)(size_t)(&Bs[1][aRow * LDT + aCol]) };

    v8f acc[2][4];
#pragma unroll
    for (int i = 0; i < 2; ++i)
#pragma unroll
        for (int j = 0; j < 4; ++j) acc[i][j] = {};

    // Async stage K-tile kt into LDS buffer buf (no VGPR data, no VALU).
    // INST_OFFSET is added to both the global and the LDS address, so the
    // second 16B chunk of each 32B row-piece is just "offset:16".
    auto stage = [&](int kt, int buf) {
        const _Float16* aS = Xrow + kt * BK;
        const _Float16* bS = Wrow + kt * BK;
        asm volatile("global_load_async_to_lds_b128 %0, %1, off"
                     :: "v"(dA[buf]), "v"(aS) : "memory");
        asm volatile("global_load_async_to_lds_b128 %0, %1, off offset:16"
                     :: "v"(dA[buf]), "v"(aS) : "memory");
        asm volatile("global_load_async_to_lds_b128 %0, %1, off"
                     :: "v"(dB[buf]), "v"(bS) : "memory");
        asm volatile("global_load_async_to_lds_b128 %0, %1, off offset:16"
                     :: "v"(dB[buf]), "v"(bS) : "memory");
    };

    stage(0, 0);
    asm volatile("s_wait_asynccnt 0" ::: "memory");
    __syncthreads();

    const int c1 = (lane >> 4) << 3;   // A K-chunk select per ISA A-layout
    const int kh = (lane >> 4) << 3;   // B K-chunk select per ISA A-layout

#pragma unroll 2
    for (int kt = 0; kt < NKT; ++kt) {
        const int buf = kt & 1;
        if (kt + 1 < NKT) stage(kt + 1, buf ^ 1);

        Frag a[2], b[4];
#pragma unroll
        for (int tm = 0; tm < 2; ++tm) {
            const int r = (wm * 32 + tm * 16 + (lane & 15)) * LDT;
            a[tm].q[0] = *(const uint4*)&As[buf][r + c1];
            a[tm].q[1] = *(const uint4*)&As[buf][r + c1 + 16];
        }
#pragma unroll
        for (int tn = 0; tn < 4; ++tn) {
            const int r = (wn * 64 + tn * 16 + (lane & 15)) * LDT;
            b[tn].q[0] = *(const uint4*)&Bs[buf][r + kh];
            b[tn].q[1] = *(const uint4*)&Bs[buf][r + kh + 16];
        }
#pragma unroll
        for (int tm = 0; tm < 2; ++tm)
#pragma unroll
            for (int tn = 0; tn < 4; ++tn)
                acc[tm][tn] = __builtin_amdgcn_wmma_f32_16x16x32_f16(
                    false, a[tm].v, false, b[tn].v, (short)0, acc[tm][tn], false, false);

        // ASYNCcnt is NOT covered by __syncthreads(); drain it before the
        // barrier so the next iteration's ds_loads see the staged tile.
        asm volatile("s_wait_asynccnt 0" ::: "memory");
        __syncthreads();
    }

    // Fused LIF epilogue (each (row,col) owned by exactly one lane/VGPR)
    const int rOff = (lane >> 4) << 3;      // C/D layout: lanes>=16 hold M=r+8
#pragma unroll
    for (int tm = 0; tm < 2; ++tm) {
#pragma unroll
        for (int tn = 0; tn < 4; ++tn) {
            const int col = nBase + wn * 64 + tn * 16 + (lane & 15);
            const int row0 = mBase + wm * 32 + tm * 16 + rOff;
            const float bias = b1p[col];
#pragma unroll
            for (int r = 0; r < 8; ++r) {
                const size_t idx = (size_t)(row0 + r) * HP + col;
                const float mo    = mem1[idx];
                const float cur   = acc[tm][tn][r] + bias;
                const float reset = (mo > THR) ? THR : 0.0f;     // detached reset
                const float mn    = BETA * mo + cur - reset;
                mem1[idx] = mn;
                spk1[idx] = (_Float16)((mn > THR) ? 1.0f : 0.0f);
            }
        }
    }
}

// ---------------------------------------------------------------------------
// GEMM2 + LIF layer 2 : cur2 = spk1 @ W2^T ; writes spk2/mem2 into d_out
//   one wave per 16-row M tile (256 waves total), K = 1024, N padded to 16
// ---------------------------------------------------------------------------
__global__ __launch_bounds__(256)
void gemm2_lif_kernel(const _Float16* __restrict__ spk1,    // [B, HP]
                      const _Float16* __restrict__ W2h,     // [16, HP]
                      const float* __restrict__ b2,         // [NOUT]
                      float* __restrict__ mem2,             // [B, 16]
                      float* __restrict__ outSpk,           // [B, NOUT] (step t)
                      float* __restrict__ outMem)           // [B, NOUT] (step t)
{
    const int lane = threadIdx.x & 31;
    const int w    = blockIdx.x * 8 + (threadIdx.x >> 5);   // 0..255
    const int rowBase = w << 4;

    v8f acc = {};
    const _Float16* aptr = spk1 + (size_t)(rowBase + (lane & 15)) * HP + ((lane >> 4) << 3);
    const _Float16* bptr = W2h  + (size_t)(lane & 15) * HP + ((lane >> 4) << 3);

#pragma unroll 4
    for (int kt = 0; kt < HP / 32; ++kt) {
        Frag a, b;
        const int k = kt << 5;
        a.q[0] = *(const uint4*)(aptr + k);
        a.q[1] = *(const uint4*)(aptr + k + 16);
        b.q[0] = *(const uint4*)(bptr + k);
        b.q[1] = *(const uint4*)(bptr + k + 16);
        acc = __builtin_amdgcn_wmma_f32_16x16x32_f16(
            false, a.v, false, b.v, (short)0, acc, false, false);
    }

    const int col = lane & 15;
    if (col < NOUT) {
        const int rOff = (lane >> 4) << 3;
        const float bias = b2[col];
#pragma unroll
        for (int r = 0; r < 8; ++r) {
            const int row = rowBase + rOff + r;
            const int idx = (row << 4) + col;
            const float mo    = mem2[idx];
            const float cur   = acc[r] + bias;
            const float reset = (mo > THR) ? THR : 0.0f;
            const float mn    = BETA * mo + cur - reset;
            mem2[idx] = mn;
            outSpk[row * NOUT + col] = (mn > THR) ? 1.0f : 0.0f;
            outMem[row * NOUT + col] = mn;
        }
    }
}

// ---------------------------------------------------------------------------
// Launch
// ---------------------------------------------------------------------------
extern "C" void kernel_launch(void* const* d_in, const int* in_sizes, int n_in,
                              void* d_out, int out_size, void* d_ws, size_t ws_size,
                              hipStream_t stream) {
    (void)in_sizes; (void)n_in; (void)out_size; (void)ws_size;
    const float* x  = (const float*)d_in[0];   // [25, 4096, 784]
    const float* W1 = (const float*)d_in[1];   // [1000, 784]
    const float* b1 = (const float*)d_in[2];   // [1000]
    const float* W2 = (const float*)d_in[3];   // [10, 1000]
    const float* b2 = (const float*)d_in[4];   // [10]
    float* out = (float*)d_out;                // spk_rec | mem_rec, [25,4096,10] each

    // Workspace carve-up (~34 MB)
    float*     mem1 = (float*)d_ws;                          // B*HP f32
    float*     mem2 = mem1 + (size_t)B * HP;                 // B*16 f32
    _Float16*  spk1 = (_Float16*)(mem2 + (size_t)B * OP16);  // B*HP f16
    _Float16*  W1h  = spk1 + (size_t)B * HP;                 // HP*KP f16
    _Float16*  W2h  = W1h + (size_t)HP * KP;                 // 16*HP f16
    float*     b1p  = (float*)(W2h + (size_t)OP16 * HP);     // HP f32
    _Float16*  Xh   = (_Float16*)(b1p + HP);                 // B*KP f16

    const int nZero = B * HP + B * OP16;
    zero_f32_kernel<<<(nZero + 255) / 256, 256, 0, stream>>>(mem1, nZero);
    prep_w1_kernel<<<(HP * KP + 255) / 256, 256, 0, stream>>>(W1, W1h);
    prep_w2_kernel<<<(OP16 * HP + 255) / 256, 256, 0, stream>>>(W2, W2h);
    prep_b1_kernel<<<(HP + 255) / 256, 256, 0, stream>>>(b1, b1p);

    for (int t = 0; t < NSTEP; ++t) {
        const float* xt = x + (size_t)t * B * NIN;
        prep_x_kernel<<<(B * (KP / 4) + 255) / 256, 256, 0, stream>>>(xt, Xh);
        gemm1_lif_kernel<<<dim3(HP / BN, B / BM), 256, 0, stream>>>(Xh, W1h, b1p, mem1, spk1);
        gemm2_lif_kernel<<<(B / 16) / 8, 256, 0, stream>>>(
            spk1, W2h, b2, mem2,
            out + (size_t)t * B * NOUT,
            out + (size_t)NSTEP * B * NOUT + (size_t)t * B * NOUT);
    }
}
